// FFTPermeabilityPredictorPatchPhysics_36962488550072
// MI455X (gfx1250) — matrix operand order repaired and checked by reading, not compiled, not run on hardware
//
#include <hip/hip_runtime.h>
#include <math.h>

// ---------------------------------------------------------------------------
// FFT-mixer ViT forward for MI455X (gfx1250).
// ~360 GFLOP, working set L2-resident (192MB) -> compute bound on MLP GEMMs.
// GEMMs: v_wmma_f32_16x16x32_f16, 32x64 wave macro-tiles (2x4 WMMA).
// Weights pre-transposed+converted to f16 [N,K] once per launch so BOTH
// operand fragments are contiguous b128 loads (no scalar gathers, no cvt in
// the hot loop). f16-only activation buffers where only WMMA consumes them.
// 48-pt rfft/irfft as VALU DFT with LDS twiddles. TDM prefetch where avail.
// ---------------------------------------------------------------------------

namespace {

constexpr int kB = 64, kD = 384, kH = 8, kHD = 48, kFB = 25, kS = 197, kL = 12,
              kNP = 196, kDC = 128, kHP = 14;
constexpr int kMT = kB * kS;    // 12608 token rows
constexpr int kMP = kB * kNP;   // 12544 patch rows

typedef __attribute__((ext_vector_type(16))) _Float16 v16h;
typedef __attribute__((ext_vector_type(8)))  _Float16 v8h;
typedef __attribute__((ext_vector_type(8)))  float    v8f;

__device__ __forceinline__ float gelu_f(float x) {
  return 0.5f * x * (1.0f + erff(x * 0.70710678118654752440f));
}

struct f8 { float v[8]; };

__device__ __forceinline__ f8 load8f(const float* __restrict__ p) {
  f8 r;
  const float4 q0 = *(const float4*)p;
  const float4 q1 = *(const float4*)(p + 4);
  r.v[0] = q0.x; r.v[1] = q0.y; r.v[2] = q0.z; r.v[3] = q0.w;
  r.v[4] = q1.x; r.v[5] = q1.y; r.v[6] = q1.z; r.v[7] = q1.w;
  return r;
}
__device__ __forceinline__ v8h cvt8h(f8 q) {
  v8h r;
  #pragma unroll
  for (int t = 0; t < 8; ++t) r[t] = (_Float16)q.v[t];
  return r;
}
__device__ __forceinline__ v16h cat16(v8h lo, v8h hi) {
  return __builtin_shufflevector(lo, hi, 0, 1, 2, 3, 4, 5, 6, 7, 8, 9, 10, 11, 12,
                                 13, 14, 15);
}
__device__ __forceinline__ v8h  load8h(const _Float16* __restrict__ p) {
  return *(const v8h*)p;
}
__device__ __forceinline__ v16h load16h(const _Float16* __restrict__ p) {
  return cat16(*(const v8h*)p, *(const v8h*)(p + 8));
}

// ------------------- optional TDM prefetch (async tensor path) --------------
#if defined(__has_builtin)
#if __has_builtin(__builtin_amdgcn_tensor_load_to_lds)
#define HAVE_TDM 1
#endif
#endif
#ifndef HAVE_TDM
#define HAVE_TDM 0
#endif

#if HAVE_TDM
typedef unsigned int u32x4 __attribute__((ext_vector_type(4)));
typedef int i32x8 __attribute__((ext_vector_type(8)));
typedef int i32x4 __attribute__((ext_vector_type(4)));

// TDM load of a [rows x 32] f16 tile (row stride = rowlen elts) into LDS 0.
// Pure L2-warming prefetch; landing zone never read, results unaffected.
__device__ __forceinline__ void tdm_prefetch(const void* gaddr, int rows, int rowlen) {
  unsigned long long ga = (unsigned long long)gaddr;
  u32x4 g0;
  g0[0] = 1u;                                         // count=1 (valid), user mode
  g0[1] = 0u;                                         // lds_addr = 0
  g0[2] = (unsigned)(ga & 0xffffffffu);               // global_addr[31:0]
  g0[3] = (unsigned)((ga >> 32) & 0x01ffffffu) | (2u << 30);  // addr[56:32], type=2
  i32x8 g1;
  g1[0] = (1 << 16);                                  // data_size=2B; no mask/flags
  g1[1] = (rowlen & 0xffff) << 16;                    // tensor_dim0[15:0]
  g1[2] = (rowlen >> 16) | ((rows & 0xffff) << 16);   // dim0[31:16] | dim1[15:0]
  g1[3] = (rows >> 16) | (32 << 16);                  // dim1[31:16] | tile_dim0=32
  g1[4] = rows & 0xffff;                              // tile_dim1=rows, tile_dim2=0
  g1[5] = rowlen;                                     // tensor_dim0_stride[31:0]
  g1[6] = 0;
  g1[7] = 0;
  i32x4 z4 = {0, 0, 0, 0};
#if defined(__clang_major__) && __clang_major__ >= 23
  i32x8 z8 = {0, 0, 0, 0, 0, 0, 0, 0};
  __builtin_amdgcn_tensor_load_to_lds(g0, g1, z4, z4, z8, 0);
#else
  __builtin_amdgcn_tensor_load_to_lds(g0, g1, z4, z4, 0);
#endif
}
#endif  // HAVE_TDM

// -------- one wave computes an (MB*16) x (NB*16) macro-tile ------------------
// Fragment layouts used here (wave32; A and B share one K order):
//   A 16x32 f16: M = lane%16, half selects K range, slots = consecutive K
//   B 32x16 f16: N = lane%16, half selects K range, slots = consecutive K
//   C/D: N = lane%16, VGPR v -> M = half*8 + v
// fa8(row, kb) -> 8 halfs at K=kb..kb+7 ; fb16(col, kb) -> 16 halfs K=kb..kb+15
template <int MB, int NB, class FA, class FB, class FC>
__device__ __forceinline__ void wmma_block(int M0, int N0, int K, FA fa8, FB fb16,
                                           FC fc) {
  const int lane = threadIdx.x & 31;
  const int half = lane >> 4;
  const int l16  = lane & 15;
  v8f acc[MB][NB];
  #pragma unroll
  for (int i = 0; i < MB; ++i)
    #pragma unroll
    for (int j = 0; j < NB; ++j)
      acc[i][j] = v8f{0.f, 0.f, 0.f, 0.f, 0.f, 0.f, 0.f, 0.f};
  for (int k0 = 0; k0 < K; k0 += 32) {
    v16h a[MB];
    #pragma unroll
    for (int i = 0; i < MB; ++i) {
      const int arow = M0 + i * 16 + l16;
      a[i] = cat16(fa8(arow, k0 + half * 16), fa8(arow, k0 + half * 16 + 8));
    }
    v16h b[NB];
    #pragma unroll
    for (int j = 0; j < NB; ++j)
      b[j] = fb16(N0 + j * 16 + l16, k0 + half * 16);
    #pragma unroll
    for (int i = 0; i < MB; ++i)
      #pragma unroll
      for (int j = 0; j < NB; ++j)
        acc[i][j] = __builtin_amdgcn_wmma_f32_16x16x32_f16(
            false, a[i], false, b[j], (short)0, acc[i][j], false, false);
  }
  #pragma unroll
  for (int i = 0; i < MB; ++i)
    #pragma unroll
    for (int j = 0; j < NB; ++j)
      #pragma unroll
      for (int v = 0; v < 8; ++v)
        fc(M0 + i * 16 + half * 8 + v, N0 + j * 16 + l16, acc[i][j][v]);
}

// -------- weight transpose+convert: W[K,N] f32 -> Wt[N,K] f16 ---------------
// blockIdx.z selects the matrix (layer); LDS-tiled for coalescing both ways.
__global__ void transpose_h_kernel(const float* __restrict__ W,
                                   _Float16* __restrict__ Wt, int K, int N) {
  __shared__ float t[32][33];
  const float* Wl = W + (size_t)blockIdx.z * K * N;
  _Float16* Wtl   = Wt + (size_t)blockIdx.z * K * N;
  const int kb = blockIdx.y * 32, nb = blockIdx.x * 32;
  for (int yy = threadIdx.y; yy < 32; yy += 8) {
    int k = kb + yy, n = nb + threadIdx.x;
    t[yy][threadIdx.x] = (k < K && n < N) ? Wl[(size_t)k * N + n] : 0.f;
  }
  __syncthreads();
  for (int yy = threadIdx.y; yy < 32; yy += 8) {
    int n = nb + yy, k = kb + threadIdx.x;
    if (n < N && k < K) Wtl[(size_t)n * K + k] = (_Float16)t[threadIdx.x][yy];
  }
}

// Wave-parallel layernorm over D=384 (12 elems/lane), in-register.
__device__ __forceinline__ void wave_ln(float (&v)[12], int lane,
                                        const float* __restrict__ g,
                                        const float* __restrict__ b) {
  float s = 0.f;
  #pragma unroll
  for (int i = 0; i < 12; ++i) s += v[i];
  #pragma unroll
  for (int o = 16; o > 0; o >>= 1) s += __shfl_xor(s, o, 32);
  const float m = s * (1.0f / kD);
  float vs = 0.f;
  #pragma unroll
  for (int i = 0; i < 12; ++i) { float d = v[i] - m; vs += d * d; }
  #pragma unroll
  for (int o = 16; o > 0; o >>= 1) vs += __shfl_xor(vs, o, 32);
  const float r = rsqrtf(vs * (1.0f / kD) + 1e-5f);
  #pragma unroll
  for (int i = 0; i < 12; ++i) {
    int d = lane + i * 32;
    v[i] = (v[i] - m) * r * g[d] + b[d];
  }
}

// -------- stage 1: patch embed (per-channel conv == patch-flatten GEMM) ------
__global__ void patch_gemm_kernel(const float* __restrict__ x,
                                  const _Float16* __restrict__ pe_wt,
                                  const float* __restrict__ pe_b,
                                  float* __restrict__ x_img) {
  const int c = blockIdx.y;
  const int job = blockIdx.x * 8 + (threadIdx.x >> 5);
  const int njn = kDC / 64;                      // 2
  if (job >= (kMP / 32) * njn) return;           // wave-uniform
  const int mj = job / njn, nj = job % njn;
  const _Float16* Wt = pe_wt + (size_t)c * kDC * 256;  // [128 x 256] f16
  __builtin_prefetch(Wt, 0, 1);
  wmma_block<2, 4>(mj * 32, nj * 64, 256,
    [&](int row, int kb) -> v8h {
      int b = row / kNP, p = row % kNP;
      int ph = p / kHP, pw = p % kHP;
      int i = kb >> 4, j0 = kb & 15;             // 8-group stays inside one image row
      return cvt8h(load8f(
          x + ((size_t)(b * 3 + c) * 224 + ph * 16 + i) * 224 + pw * 16 + j0));
    },
    [&](int col, int kb) -> v16h { return load16h(Wt + (size_t)col * 256 + kb); },
    [&](int m, int n, float v) {
      x_img[(size_t)m * kD + c * kDC + n] = v + pe_b[c * kDC + n];
    });
}

// -------- stage 2: LN(x_img) + physics embed (+f16 mirrors) ------------------
__global__ void ln_img_phys_kernel(float* __restrict__ x_img,
                                   const float* __restrict__ pf,
                                   const float* __restrict__ phys_w,
                                   const float* __restrict__ phys_b,
                                   const float* __restrict__ g,
                                   const float* __restrict__ bb,
                                   float* __restrict__ x_phys,
                                   _Float16* __restrict__ xi_h,
                                   _Float16* __restrict__ xp_h) {
  const int row = blockIdx.x * 8 + (threadIdx.x >> 5);
  if (row >= kMP) return;
  const int lane = threadIdx.x & 31;
  float v[12];
  #pragma unroll
  for (int i = 0; i < 12; ++i) v[i] = x_img[(size_t)row * kD + lane + i * 32];
  wave_ln(v, lane, g, bb);
  const float f0 = pf[row * 6 + 0], f1 = pf[row * 6 + 1], f2 = pf[row * 6 + 2];
  const float f3 = pf[row * 6 + 3], f4 = pf[row * 6 + 4], f5 = pf[row * 6 + 5];
  #pragma unroll
  for (int i = 0; i < 12; ++i) {
    int d = lane + i * 32;
    float xp = f0 * phys_w[0 * kD + d] + f1 * phys_w[1 * kD + d] +
               f2 * phys_w[2 * kD + d] + f3 * phys_w[3 * kD + d] +
               f4 * phys_w[4 * kD + d] + f5 * phys_w[5 * kD + d] + phys_b[d];
    x_img[(size_t)row * kD + d] = v[i];
    x_phys[(size_t)row * kD + d] = xp;
    xi_h[(size_t)row * kD + d] = (_Float16)v[i];
    xp_h[(size_t)row * kD + d] = (_Float16)xp;
  }
}

// -------- stage 3: gate GEMM + fused sigmoid/mix/pos-embed -------------------
__global__ void gate_gemm_kernel(const _Float16* __restrict__ xi_h,
                                 const _Float16* __restrict__ xp_h,
                                 const float* __restrict__ x_img,
                                 const float* __restrict__ x_phys,
                                 const _Float16* __restrict__ gate_wt,
                                 const float* __restrict__ gate_b,
                                 const float* __restrict__ pos,
                                 float* __restrict__ h) {
  const int job = blockIdx.x * 8 + (threadIdx.x >> 5);
  const int njn = kD / 64;                       // 6
  if (job >= (kMP / 32) * njn) return;
  const int mj = job / njn, nj = job % njn;
  __builtin_prefetch(gate_wt, 0, 1);
  wmma_block<2, 4>(mj * 32, nj * 64, 2 * kD,
    [&](int row, int kb) -> v8h {                // kb mult of 8; 384%8==0, no straddle
      return (kb < kD) ? load8h(xi_h + (size_t)row * kD + kb)
                       : load8h(xp_h + (size_t)row * kD + kb - kD);
    },
    [&](int col, int kb) -> v16h {
      return load16h(gate_wt + (size_t)col * (2 * kD) + kb);
    },
    [&](int m, int n, float v) {
      float gt = 1.0f / (1.0f + expf(-(v + gate_b[n])));
      float xi = x_img[(size_t)m * kD + n], xp = x_phys[(size_t)m * kD + n];
      int b = m / kNP, p = m % kNP;
      h[(size_t)(b * kS + 1 + p) * kD + n] =
          gt * xi + (1.0f - gt) * xp + pos[(size_t)(1 + p) * kD + n];
    });
}

__global__ void cls_init_kernel(const float* __restrict__ cls_tok,
                                const float* __restrict__ pos,
                                float* __restrict__ h) {
  int i = blockIdx.x * blockDim.x + threadIdx.x;
  if (i >= kB * kD) return;
  int b = i / kD, d = i % kD;
  h[(size_t)(b * kS) * kD + d] = cls_tok[d] + pos[d];
}

// -------- layernorm (optionally double), writes f32 + f16 mirror -------------
__global__ void ln_kernel(const float* __restrict__ in, float* __restrict__ outp,
                          _Float16* __restrict__ outh,
                          const float* __restrict__ g1, const float* __restrict__ b1,
                          const float* __restrict__ g2, const float* __restrict__ b2,
                          int nrows, int twice) {
  const int row = blockIdx.x * 8 + (threadIdx.x >> 5);
  if (row >= nrows) return;
  const int lane = threadIdx.x & 31;
  float v[12];
  #pragma unroll
  for (int i = 0; i < 12; ++i) v[i] = in[(size_t)row * kD + lane + i * 32];
  wave_ln(v, lane, g1, b1);
  if (twice) wave_ln(v, lane, g2, b2);
  #pragma unroll
  for (int i = 0; i < 12; ++i) {
    outp[(size_t)row * kD + lane + i * 32] = v[i];
    outh[(size_t)row * kD + lane + i * 32] = (_Float16)v[i];
  }
}

__global__ void token_mean_kernel(const float* __restrict__ hn,
                                  _Float16* __restrict__ hmean_h) {
  int i = blockIdx.x * blockDim.x + threadIdx.x;
  if (i >= kB * kD) return;
  int b = i / kD, d = i % kD;
  float s = 0.f;
  for (int t = 0; t < kS; ++t) s += hn[(size_t)(b * kS + t) * kD + d];
  hmean_h[i] = (_Float16)(s * (1.0f / kS));
}

// -------- generic f16 WMMA GEMM: MODE 0 store, 1 gelu+store, 2 residual-add --
// OH: 1 -> store output as f16, 0 -> f32 (MODE 2 is always f32 accum buffer).
template <int MODE, int OH, int MB, int NB>
__global__ void gemm_h_kernel(const _Float16* __restrict__ A,
                              const _Float16* __restrict__ Wt,
                              const float* __restrict__ bias, void* __restrict__ outp,
                              int K, int N, int total_jobs) {
#if HAVE_TDM
  __shared__ float tdm_buf[1024];                // TDM scratch landing zone (4KB)
  if (threadIdx.x == 0) tdm_buf[0] = 0.0f;       // keep allocation alive
#endif
  const int job = blockIdx.x * 8 + (threadIdx.x >> 5);
  if (job >= total_jobs) return;
  const int njn = N / (16 * NB);
  const int mj = job / njn, nj = job % njn;
#if HAVE_TDM
  if ((threadIdx.x >> 5) == 0)                   // one TDM issue per block: warm L2
    tdm_prefetch(A + (size_t)(mj * 16 * MB) * K, 16 * MB, K);
#endif
  __builtin_prefetch(Wt + (size_t)(nj * 16 * NB) * K, 0, 1);
  float* outf = (float*)outp;
  _Float16* outh = (_Float16*)outp;
  wmma_block<MB, NB>(mj * 16 * MB, nj * 16 * NB, K,
    [&](int row, int kb) -> v8h { return load8h(A + (size_t)row * K + kb); },
    [&](int col, int kb) -> v16h { return load16h(Wt + (size_t)col * K + kb); },
    [&](int m, int n, float v) {
      v += bias[n];
      if (MODE == 1) v = gelu_f(v);
      if (MODE == 2)      outf[(size_t)m * N + n] += v;
      else if (OH)        outh[(size_t)m * N + n] = (_Float16)v;
      else                outf[(size_t)m * N + n] = v;
    });
#if HAVE_TDM
  if ((threadIdx.x >> 5) == 0) __builtin_amdgcn_s_wait_tensorcnt(0);
#endif
}

// -------- FFT token mixing: rfft(48)->modulate->gelu->irfft, + residual ------
__global__ void fft_attn_kernel(const float* __restrict__ hn,
                                const float* __restrict__ ap,
                                const float* __restrict__ bf,
                                const float* __restrict__ bb,
                                float* __restrict__ h) {
  __shared__ float ct[kFB][kHD];
  __shared__ float st[kFB][kHD];
  for (int i = threadIdx.x; i < kFB * kHD; i += blockDim.x) {
    int f = i / kHD, n = i % kHD;
    float ang = 6.28318530717958647692f * (float)(f * n) / 48.0f;
    ct[f][n] = cosf(ang);
    st[f][n] = sinf(ang);
  }
  __syncthreads();
  const int r = blockIdx.x * blockDim.x + threadIdx.x;
  if (r >= kB * kS * kH) return;
  const int b = r / (kS * kH);
  const int rem = r % (kS * kH);
  const int s = rem / kH, hh = rem % kH;
  const float* vin = hn + (size_t)(b * kS + s) * kD + hh * kHD;
  float v[kHD];
  #pragma unroll
  for (int n = 0; n < kHD; ++n) v[n] = vin[n];
  const float inv = 0.14433756729740644113f;  // 1/sqrt(48), ortho norm
  float fr[kFB], fi[kFB];
  for (int f = 0; f < kFB; ++f) {
    float sr = 0.f, si = 0.f;
    #pragma unroll
    for (int n = 0; n < kHD; ++n) { sr += v[n] * ct[f][n]; si -= v[n] * st[f][n]; }
    float a0 = ap[(size_t)b * (kH * kFB * 2) + (hh * kFB + f) * 2 + 0];
    float a1 = ap[(size_t)b * (kH * kFB * 2) + (hh * kFB + f) * 2 + 1];
    float eff = bf[hh * kFB + f] * (1.0f + a0);
    float efb = bb[hh * kFB + f] + a1;
    fr[f] = gelu_f(sr * inv * eff + efb);   // real bias adds to real part only
    fi[f] = gelu_f(si * inv * eff);
  }
  float* hout = h + (size_t)(b * kS + s) * kD + hh * kHD;
  for (int n = 0; n < kHD; ++n) {
    float acc = fr[0] + fr[24] * ct[24][n];  // DC + Nyquist (imag ignored by irfft)
    #pragma unroll
    for (int f = 1; f < 24; ++f) acc += 2.0f * (fr[f] * ct[f][n] - fi[f] * st[f][n]);
    hout[n] += acc * inv;
  }
}

// -------- final head ---------------------------------------------------------
__global__ void head_kernel(const float* __restrict__ h,
                            const float* __restrict__ ng, const float* __restrict__ nb,
                            const float* __restrict__ w1, const float* __restrict__ b1,
                            const float* __restrict__ w2, const float* __restrict__ b2,
                            float* __restrict__ out) {
  __shared__ float cls[kD];
  __shared__ float y[kD];
  __shared__ float red[2];
  const int b = blockIdx.x, t = threadIdx.x;
  float v = h[(size_t)(b * kS) * kD + t];
  cls[t] = v;
  __syncthreads();
  if (t == 0) {
    float s = 0.f, ss = 0.f;
    for (int i = 0; i < kD; ++i) s += cls[i];
    float m = s / kD;
    for (int i = 0; i < kD; ++i) { float d = cls[i] - m; ss += d * d; }
    red[0] = m;
    red[1] = rsqrtf(ss / kD + 1e-5f);
  }
  __syncthreads();
  cls[t] = (v - red[0]) * red[1] * ng[t] + nb[t];
  __syncthreads();
  float acc = b1[t];
  for (int k = 0; k < kD; ++k) acc += cls[k] * w1[(size_t)k * kD + t];
  y[t] = gelu_f(acc) * w2[t];
  __syncthreads();
  if (t == 0) {
    float s = b2[0];
    for (int i = 0; i < kD; ++i) s += y[i];
    out[b] = s;
  }
}

}  // namespace

extern "C" void kernel_launch(void* const* d_in, const int* in_sizes, int n_in,
                              void* d_out, int out_size, void* d_ws, size_t ws_size,
                              hipStream_t stream) {
  const float* x          = (const float*)d_in[0];
  const float* patch_f    = (const float*)d_in[1];
  const float* pe_w       = (const float*)d_in[2];
  const float* pe_b       = (const float*)d_in[3];
  const float* pe_ln_g    = (const float*)d_in[4];
  const float* pe_ln_b    = (const float*)d_in[5];
  const float* phys_w     = (const float*)d_in[6];
  const float* phys_b     = (const float*)d_in[7];
  const float* gate_w     = (const float*)d_in[8];
  const float* gate_b     = (const float*)d_in[9];
  const float* cls_tok    = (const float*)d_in[10];
  const float* pos        = (const float*)d_in[11];
  const float* ln1_g      = (const float*)d_in[12];
  const float* ln1_b      = (const float*)d_in[13];
  const float* pre_g      = (const float*)d_in[14];
  const float* pre_b      = (const float*)d_in[15];
  const float* base_filt  = (const float*)d_in[16];
  const float* base_bias  = (const float*)d_in[17];
  const float* amlp_w1    = (const float*)d_in[18];
  const float* amlp_b1    = (const float*)d_in[19];
  const float* amlp_w2    = (const float*)d_in[20];
  const float* amlp_b2    = (const float*)d_in[21];
  const float* ln2_g      = (const float*)d_in[22];
  const float* ln2_b      = (const float*)d_in[23];
  const float* mlp_w1     = (const float*)d_in[24];
  const float* mlp_b1     = (const float*)d_in[25];
  const float* mlp_w2     = (const float*)d_in[26];
  const float* mlp_b2     = (const float*)d_in[27];
  const float* norm_g     = (const float*)d_in[28];
  const float* norm_b     = (const float*)d_in[29];
  const float* head_w1    = (const float*)d_in[30];
  const float* head_b1    = (const float*)d_in[31];
  const float* head_w2    = (const float*)d_in[32];
  const float* head_b2    = (const float*)d_in[33];
  float* out = (float*)d_out;

  // ---- workspace carve-up: f32 region then f16 region (~181MB total) -------
  float* ws     = (float*)d_ws;
  float* x_img  = ws;                             // 12544*384 f32
  float* x_phys = x_img  + (size_t)kMP * kD;
  float* hbuf   = x_phys + (size_t)kMP * kD;      // 12608*384 f32
  float* hn     = hbuf   + (size_t)kMT * kD;
  float* apbuf  = hn     + (size_t)kMT * kD;      // 64*400 f32
  _Float16* hp  = (_Float16*)(apbuf + (size_t)kB * kH * kFB * 2);
  _Float16* xi_h  = hp;                           // 12544*384 f16
  _Float16* xp_h  = xi_h  + (size_t)kMP * kD;
  _Float16* hn_h  = xp_h  + (size_t)kMP * kD;     // 12608*384 f16
  _Float16* tb_h  = hn_h  + (size_t)kMT * kD;     // 12608*1536 f16
  _Float16* hm_h  = tb_h  + (size_t)kMT * 4 * kD; // 64*384 f16
  _Float16* ah_h  = hm_h  + (size_t)kB * kD;      // 64*384 f16
  _Float16* pe_wt = ah_h  + (size_t)kB * kD;      // 3*128*256 f16 [N,K]
  _Float16* gw_t  = pe_wt + (size_t)3 * kDC * 256;        // 384*768
  _Float16* aw1t  = gw_t  + (size_t)kD * 2 * kD;          // 12*384*384
  _Float16* aw2t  = aw1t  + (size_t)kL * kD * kD;         // 12*400*384
  _Float16* mw1t  = aw2t  + (size_t)kL * (kH * kFB * 2) * kD;  // 12*1536*384
  _Float16* mw2t  = mw1t  + (size_t)kL * kD * 4 * kD;     // 12*384*1536

  const dim3 blk(256);
  const dim3 tblk(32, 8);

  // ---- one-time (per launch) weight transpose+convert to f16 [N,K] ---------
  transpose_h_kernel<<<dim3(4, 8, 3), tblk, 0, stream>>>(pe_w, pe_wt, 256, kDC);
  transpose_h_kernel<<<dim3(12, 24, 1), tblk, 0, stream>>>(gate_w, gw_t, 2 * kD, kD);
  transpose_h_kernel<<<dim3(12, 12, kL), tblk, 0, stream>>>(amlp_w1, aw1t, kD, kD);
  transpose_h_kernel<<<dim3(13, 12, kL), tblk, 0, stream>>>(amlp_w2, aw2t, kD,
                                                            kH * kFB * 2);
  transpose_h_kernel<<<dim3(48, 12, kL), tblk, 0, stream>>>(mlp_w1, mw1t, kD, 4 * kD);
  transpose_h_kernel<<<dim3(12, 48, kL), tblk, 0, stream>>>(mlp_w2, mw2t, 4 * kD, kD);

  // ---- patch embed / physics / gate / cls -----------------------------------
  patch_gemm_kernel<<<dim3((kMP / 32) * (kDC / 64) / 8, 3), blk, 0, stream>>>(
      x, pe_wt, pe_b, x_img);
  ln_img_phys_kernel<<<dim3((kMP + 7) / 8), blk, 0, stream>>>(
      x_img, patch_f, phys_w, phys_b, pe_ln_g, pe_ln_b, x_phys, xi_h, xp_h);
  gate_gemm_kernel<<<dim3((kMP / 32) * (kD / 64) / 8), blk, 0, stream>>>(
      xi_h, xp_h, x_img, x_phys, gw_t, gate_b, pos, hbuf);
  cls_init_kernel<<<dim3((kB * kD + 255) / 256), blk, 0, stream>>>(cls_tok, pos, hbuf);

  for (int l = 0; l < kL; ++l) {
    // hn = LN(LN(h, ln1), pre)
    ln_kernel<<<dim3((kMT + 7) / 8), blk, 0, stream>>>(
        hbuf, hn, hn_h, ln1_g + l * kD, ln1_b + l * kD, pre_g + l * kD, pre_b + l * kD,
        kMT, 1);
    token_mean_kernel<<<dim3((kB * kD + 255) / 256), blk, 0, stream>>>(hn, hm_h);
    // adaptive MLP: [64,384]@[384,384] gelu -> f16, then @[384,400] -> f32
    {
      int jobs1 = (kB / 32) * (kD / 64);               // 12
      gemm_h_kernel<1, 1, 2, 4><<<dim3((jobs1 + 7) / 8), blk, 0, stream>>>(
          hm_h, aw1t + (size_t)l * kD * kD, amlp_b1 + l * kD, ah_h, kD, kD, jobs1);
      int jobs2 = (kB / 32) * (kH * kFB * 2 / 16);     // 2*25 = 50
      gemm_h_kernel<0, 0, 2, 1><<<dim3((jobs2 + 7) / 8), blk, 0, stream>>>(
          ah_h, aw2t + (size_t)l * (kH * kFB * 2) * kD, amlp_b2 + l * (kH * kFB * 2),
          apbuf, kD, kH * kFB * 2, jobs2);
    }
    // spectral mixing + residual into h
    fft_attn_kernel<<<dim3((kB * kS * kH + 255) / 256), blk, 0, stream>>>(
        hn, apbuf, base_filt + l * kH * kFB, base_bias + l * kH * kFB, hbuf);
    // MLP: h += gelu(LN(h)@W1+b1)@W2+b2  (hidden kept f16)
    ln_kernel<<<dim3((kMT + 7) / 8), blk, 0, stream>>>(
        hbuf, hn, hn_h, ln2_g + l * kD, ln2_b + l * kD, nullptr, nullptr, kMT, 0);
    {
      int jobs1 = (kMT / 32) * (4 * kD / 64);          // 394*24 = 9456
      gemm_h_kernel<1, 1, 2, 4><<<dim3((jobs1 + 7) / 8), blk, 0, stream>>>(
          hn_h, mw1t + (size_t)l * 4 * kD * kD, mlp_b1 + l * 4 * kD, tb_h,
          kD, 4 * kD, jobs1);
      int jobs2 = (kMT / 32) * (kD / 64);              // 394*6 = 2364
      gemm_h_kernel<2, 0, 2, 4><<<dim3((jobs2 + 7) / 8), blk, 0, stream>>>(
          tb_h, mw2t + (size_t)l * 4 * kD * kD, mlp_b2 + l * kD, hbuf,
          4 * kD, kD, jobs2);
    }
  }

  head_kernel<<<dim3(kB), dim3(kD), 0, stream>>>(
      hbuf, norm_g, norm_b, head_w1, head_b1, head_w2, head_b2, out);
}
